// AttentionGNN_76175539962269
// MI455X (gfx1250) — matrix operand order repaired and checked by reading, not compiled, not run on hardware
//
#include <hip/hip_runtime.h>
#include <hip/hip_bf16.h>

typedef __bf16 bhalf;
typedef __attribute__((ext_vector_type(4)))  __bf16 v4bf;
typedef __attribute__((ext_vector_type(8)))  __bf16 v8bf;
typedef __attribute__((ext_vector_type(16))) __bf16 v16bf;
typedef __attribute__((ext_vector_type(8)))  float  v8f;

// ---------------------------------------------------------------------------
// Utility kernels
// ---------------------------------------------------------------------------
__global__ void zero_u32(unsigned* __restrict__ p, long n) {
    long i = (long)blockIdx.x * blockDim.x + threadIdx.x;
    if (i < n) p[i] = 0u;
}

// n must be a multiple of 4
__global__ void zero_u32x4(uint4* __restrict__ p, long n4) {
    long i = (long)blockIdx.x * blockDim.x + threadIdx.x;
    if (i < n4) p[i] = make_uint4(0u, 0u, 0u, 0u);
}

__global__ void cast_f32_bf16(const float* __restrict__ in, bhalf* __restrict__ out, long n) {
    long i = (long)blockIdx.x * blockDim.x + threadIdx.x;
    if (i < n) out[i] = (bhalf)in[i];
}

// W is (din x dout) row-major; write Wt as (dout x din) so B fragments are
// K-contiguous per output column.
__global__ void transpose_cast(const float* __restrict__ W, bhalf* __restrict__ Wt,
                               int din, int dout) {
    long i = (long)blockIdx.x * blockDim.x + threadIdx.x;
    long total = (long)din * dout;
    if (i >= total) return;
    int k = (int)(i / dout);
    int n = (int)(i % dout);
    Wt[(size_t)n * din + k] = (bhalf)W[i];
}

// ---------------------------------------------------------------------------
// WMMA bf16 GEMM:  H[N][dout] = X[N][din] @ W   (W supplied transposed: Wt[dout][din])
// One wave computes a 16x64 output tile (4 WMMA accumulators).
// Fragment layouts per CDNA5 ISA (16-bit A 16x32; B 32x16; f32 C/D).
// ---------------------------------------------------------------------------
__device__ inline v16bf ldfrag(const bhalf* p0, const bhalf* p1) {
    v8bf lo = *(const v8bf*)p0;
    v8bf hi = *(const v8bf*)p1;
    v16bf r;
#pragma unroll
    for (int j = 0; j < 8; ++j) { r[j] = lo[j]; r[j + 8] = hi[j]; }
    return r;
}

__global__ __launch_bounds__(256) void gat_gemm_bf16(
    const bhalf* __restrict__ X, const bhalf* __restrict__ Wt,
    float* __restrict__ H, int Nn, int din, int dout)
{
    const int lane = threadIdx.x & 31;
    const int wv   = threadIdx.x >> 5;
    const int row0 = (blockIdx.x * 8 + wv) * 16;
    if (row0 >= Nn) return;                 // uniform per wave -> EXEC stays all-ones
    const int colBase = blockIdx.y * 64;
    const int mn = lane & 15;               // M (A rows) / N (B cols) within tile
    const int hh = lane >> 4;               // lane half selects K sub-ranges
    int arow = row0 + mn;
    if (arow > Nn - 1) arow = Nn - 1;       // clamp; clamped rows never stored
    const bhalf* aRow = X + (size_t)arow * din;
    const int ks = hh * 8;                  // unused; A also takes K = kb..kb+15 (same order as B)
    const int kb = hh * 16;                 // B: K = kb..kb+15
    const bhalf* bRow0 = Wt + (size_t)(colBase +  0 + mn) * din;
    const bhalf* bRow1 = Wt + (size_t)(colBase + 16 + mn) * din;
    const bhalf* bRow2 = Wt + (size_t)(colBase + 32 + mn) * din;
    const bhalf* bRow3 = Wt + (size_t)(colBase + 48 + mn) * din;

    v8f acc0 = {}, acc1 = {}, acc2 = {}, acc3 = {};
    for (int k0 = 0; k0 < din; k0 += 32) {
        v16bf a = ldfrag(aRow + k0 + kb, aRow + k0 + kb + 8);
        v16bf b0 = ldfrag(bRow0 + k0 + kb, bRow0 + k0 + kb + 8);
        acc0 = __builtin_amdgcn_wmma_f32_16x16x32_bf16(false, a, false, b0, (short)0, acc0, false, false);
        v16bf b1 = ldfrag(bRow1 + k0 + kb, bRow1 + k0 + kb + 8);
        acc1 = __builtin_amdgcn_wmma_f32_16x16x32_bf16(false, a, false, b1, (short)0, acc1, false, false);
        v16bf b2 = ldfrag(bRow2 + k0 + kb, bRow2 + k0 + kb + 8);
        acc2 = __builtin_amdgcn_wmma_f32_16x16x32_bf16(false, a, false, b2, (short)0, acc2, false, false);
        v16bf b3 = ldfrag(bRow3 + k0 + kb, bRow3 + k0 + kb + 8);
        acc3 = __builtin_amdgcn_wmma_f32_16x16x32_bf16(false, a, false, b3, (short)0, acc3, false, false);
    }

#pragma unroll
    for (int r = 0; r < 8; ++r) {
        int orow = row0 + hh * 8 + r;       // C/D layout: VGPR r, half hh -> M = r + 8*hh
        if (orow < Nn) {
            float* op = H + (size_t)orow * dout + colBase + mn;
            op[0]  = acc0[r];
            op[16] = acc1[r];
            op[32] = acc2[r];
            op[48] = acc3[r];
        }
    }
}

// ---------------------------------------------------------------------------
// alpha_s / alpha_d: one wave per (node, head); shuffle reduction (wave32)
// ---------------------------------------------------------------------------
__global__ void alpha_kernel(const float* __restrict__ H,
                             const float* __restrict__ a_src, const float* __restrict__ a_dst,
                             float* __restrict__ as_out, float* __restrict__ ad_out,
                             int Nn, int heads, int chan)
{
    int wave = (int)(((long)blockIdx.x * blockDim.x + threadIdx.x) >> 5);
    int lane = threadIdx.x & 31;
    if (wave >= Nn * heads) return;
    int n  = wave / heads;
    int hd = wave % heads;
    const float* hp  = H + (size_t)n * heads * chan + (size_t)hd * chan;
    const float* asp = a_src + (size_t)hd * chan;
    const float* adp = a_dst + (size_t)hd * chan;
    float ss = 0.f, sd = 0.f;
    for (int c = lane; c < chan; c += 32) {
        float v = hp[c];
        ss += v * asp[c];
        sd += v * adp[c];
    }
#pragma unroll
    for (int o = 16; o > 0; o >>= 1) {
        ss += __shfl_xor(ss, o, 32);
        sd += __shfl_xor(sd, o, 32);
    }
    if (lane == 0) { as_out[wave] = ss; ad_out[wave] = sd; }
}

// ---------------------------------------------------------------------------
// Edge softmax pipeline (order-preserving uint keys for float atomicMax)
// ---------------------------------------------------------------------------
__device__ inline unsigned f2key(float f) {
    unsigned u = __float_as_uint(f);
    return (u & 0x80000000u) ? ~u : (u | 0x80000000u);
}
__device__ inline float key2f(unsigned k) {
    unsigned u = (k & 0x80000000u) ? (k & 0x7FFFFFFFu) : ~k;
    return __uint_as_float(u);
}

__global__ void edge_e_max(const int* __restrict__ srcA, const int* __restrict__ dstA,
                           int Ee, int nE,
                           const float* __restrict__ as, const float* __restrict__ ad,
                           float* __restrict__ ebuf, unsigned* __restrict__ mbits, int heads)
{
    int e = blockIdx.x * blockDim.x + threadIdx.x;
    if (e >= nE) return;
    int s = (e < Ee) ? srcA[e] : (e - Ee);
    int d = (e < Ee) ? dstA[e] : (e - Ee);
    for (int hd = 0; hd < heads; ++hd) {
        float v = as[(size_t)s * heads + hd] + ad[(size_t)d * heads + hd];
        v = (v > 0.f) ? v : 0.2f * v;                       // leaky_relu 0.2
        ebuf[(size_t)e * heads + hd] = v;
        atomicMax(&mbits[(size_t)d * heads + hd], f2key(v));
    }
}

__global__ void edge_exp(const int* __restrict__ dstA, int Ee, int nE,
                         float* __restrict__ ebuf, const unsigned* __restrict__ mbits,
                         float* __restrict__ denom, int heads)
{
    int e = blockIdx.x * blockDim.x + threadIdx.x;
    if (e >= nE) return;
    int d = (e < Ee) ? dstA[e] : (e - Ee);
    for (int hd = 0; hd < heads; ++hd) {
        float m  = key2f(mbits[(size_t)d * heads + hd]);
        float ex = expf(ebuf[(size_t)e * heads + hd] - m);
        ebuf[(size_t)e * heads + hd] = ex;
        atomicAdd(&denom[(size_t)d * heads + hd], ex);
    }
}

__global__ void edge_alpha(const int* __restrict__ dstA, int Ee, int nE,
                           float* __restrict__ ebuf, const float* __restrict__ denom, int heads)
{
    long i = (long)blockIdx.x * blockDim.x + threadIdx.x;
    if (i >= (long)nE * heads) return;
    int e  = (int)(i / heads);
    int hd = (int)(i % heads);
    int d  = (e < Ee) ? dstA[e] : (e - Ee);
    ebuf[i] = ebuf[i] / (denom[(size_t)d * heads + hd] + 1e-16f);
}

// One thread per (edge, 4 channels): b128 gather of h[src], scale by alpha,
// 4x global_atomic_add_f32 scatter into agg[dst].
__global__ void aggregate4(const int* __restrict__ srcA, const int* __restrict__ dstA,
                           int Ee, int nE,
                           const float* __restrict__ H, const float* __restrict__ ebuf,
                           float* __restrict__ agg, int doShift, int chShift, int heads)
{
    const int qShift = doShift - 2;                  // quads per row
    long idx = (long)blockIdx.x * blockDim.x + threadIdx.x;
    if (idx >= ((long)nE << qShift)) return;
    int e  = (int)(idx >> qShift);
    int ch = (int)((idx & ((1 << qShift) - 1)) << 2);
    int hd = ch >> chShift;
    int s = (e < Ee) ? srcA[e] : (e - Ee);
    int d = (e < Ee) ? dstA[e] : (e - Ee);
    float alpha = ebuf[(size_t)e * heads + hd];
    const float4 hv = *(const float4*)(H + ((size_t)s << doShift) + ch);
    float* ap = agg + ((size_t)d << doShift) + ch;
    atomicAdd(ap + 0, hv.x * alpha);
    atomicAdd(ap + 1, hv.y * alpha);
    atomicAdd(ap + 2, hv.z * alpha);
    atomicAdd(ap + 3, hv.w * alpha);
}

// ---------------------------------------------------------------------------
// Epilogues (4-wide)
// ---------------------------------------------------------------------------
__global__ void bias_elu_cast4(const float* __restrict__ agg, const float* __restrict__ b,
                               bhalf* __restrict__ xout, long total4, int dout)
{
    long i = (long)blockIdx.x * blockDim.x + threadIdx.x;
    if (i >= total4) return;
    long base = i << 2;
    int j = (int)(base & (dout - 1));
    float4 v = *(const float4*)(agg + base);
    float4 bb = *(const float4*)(b + j);
    float r0 = v.x + bb.x, r1 = v.y + bb.y, r2 = v.z + bb.z, r3 = v.w + bb.w;
    r0 = (r0 > 0.f) ? r0 : (expf(r0) - 1.f);
    r1 = (r1 > 0.f) ? r1 : (expf(r1) - 1.f);
    r2 = (r2 > 0.f) ? r2 : (expf(r2) - 1.f);
    r3 = (r3 > 0.f) ? r3 : (expf(r3) - 1.f);
    v4bf o; o[0] = (bhalf)r0; o[1] = (bhalf)r1; o[2] = (bhalf)r2; o[3] = (bhalf)r3;
    *(v4bf*)(xout + base) = o;
}

__global__ void bias_out4(const float* __restrict__ agg, const float* __restrict__ b,
                          float* __restrict__ out, long total4, int dout)
{
    long i = (long)blockIdx.x * blockDim.x + threadIdx.x;
    if (i >= total4) return;
    long base = i << 2;
    int j = (int)(base & (dout - 1));
    float4 v = *(const float4*)(agg + base);
    float4 bb = *(const float4*)(b + j);
    float4 o = make_float4(v.x + bb.x, v.y + bb.y, v.z + bb.z, v.w + bb.w);
    *(float4*)(out + base) = o;
}

// ---------------------------------------------------------------------------
// Host launcher
// ---------------------------------------------------------------------------
extern "C" void kernel_launch(void* const* d_in, const int* in_sizes, int n_in,
                              void* d_out, int out_size, void* d_ws, size_t ws_size,
                              hipStream_t stream)
{
    const int INF = 128;
    const int Nn = in_sizes[0] / INF;       // 20000
    const int Ee = in_sizes[1] / 2;         // 320000
    const int nE = Ee + Nn;                 // edges + self loops

    const float* x0     = (const float*)d_in[0];
    const int*   srcArr = (const int*)d_in[1];
    const int*   dstArr = srcArr + Ee;

    const float *Wl[4], *aS[4], *aD[4], *bl[4];
    for (int l = 0; l < 4; ++l) {
        Wl[l] = (const float*)d_in[2 + 4 * l + 0];
        aS[l] = (const float*)d_in[2 + 4 * l + 1];
        aD[l] = (const float*)d_in[2 + 4 * l + 2];
        bl[l] = (const float*)d_in[2 + 4 * l + 3];
    }

    const int dinA[4]    = {128, 1024, 1024, 1024};
    const int doutA[4]   = {1024, 1024, 1024, 64};
    const int headsA[4]  = {4, 4, 4, 1};
    const int chanA[4]   = {256, 256, 256, 64};
    const int chShiftA[4]= {8, 8, 8, 6};
    const int doShiftA[4]= {10, 10, 10, 6};

    size_t off = 0;
    auto alloc = [&](size_t bytes) -> void* {
        off = (off + 255) & ~(size_t)255;
        void* p = (char*)d_ws + off;
        off += bytes;
        return p;
    };
    bhalf* xbf  = (bhalf*)alloc((size_t)Nn * 1024 * sizeof(bhalf));
    float* Hbuf = (float*)alloc((size_t)Nn * 1024 * sizeof(float));
    float* agg  = (float*)alloc((size_t)Nn * 1024 * sizeof(float));
    bhalf* Wt[4];
    for (int l = 0; l < 4; ++l)
        Wt[l] = (bhalf*)alloc((size_t)dinA[l] * doutA[l] * sizeof(bhalf));
    float*    asb   = (float*)alloc((size_t)Nn * 4 * sizeof(float));
    float*    adb   = (float*)alloc((size_t)Nn * 4 * sizeof(float));
    unsigned* mbits = (unsigned*)alloc((size_t)Nn * 4 * sizeof(unsigned));
    float*    denom = (float*)alloc((size_t)Nn * 4 * sizeof(float));
    float*    ebuf  = (float*)alloc((size_t)nE * 4 * sizeof(float));
    (void)ws_size; (void)n_in; (void)out_size;

    auto cdiv = [](long a, long b) { return (int)((a + b - 1) / b); };

    // One-time prep: cast input + cast/transpose all weights
    cast_f32_bf16<<<cdiv((long)Nn * INF, 256), 256, 0, stream>>>(x0, xbf, (long)Nn * INF);
    for (int l = 0; l < 4; ++l)
        transpose_cast<<<cdiv((long)dinA[l] * doutA[l], 256), 256, 0, stream>>>(
            Wl[l], Wt[l], dinA[l], doutA[l]);

    for (int l = 0; l < 4; ++l) {
        const int din = dinA[l], dout = doutA[l], heads = headsA[l], chan = chanA[l];

        dim3 gg((unsigned)cdiv(cdiv(Nn, 16), 8), (unsigned)(dout / 64));
        gat_gemm_bf16<<<gg, 256, 0, stream>>>(xbf, Wt[l], Hbuf, Nn, din, dout);

        alpha_kernel<<<cdiv((long)Nn * heads * 32, 256), 256, 0, stream>>>(
            Hbuf, aS[l], aD[l], asb, adb, Nn, heads, chan);

        zero_u32<<<cdiv((long)Nn * heads, 256), 256, 0, stream>>>(mbits, (long)Nn * heads);
        zero_u32<<<cdiv((long)Nn * heads, 256), 256, 0, stream>>>((unsigned*)denom, (long)Nn * heads);
        zero_u32x4<<<cdiv(((long)Nn * dout) >> 2, 256), 256, 0, stream>>>(
            (uint4*)agg, ((long)Nn * dout) >> 2);

        edge_e_max<<<cdiv(nE, 256), 256, 0, stream>>>(srcArr, dstArr, Ee, nE, asb, adb, ebuf, mbits, heads);
        edge_exp<<<cdiv(nE, 256), 256, 0, stream>>>(dstArr, Ee, nE, ebuf, mbits, denom, heads);
        edge_alpha<<<cdiv((long)nE * heads, 256), 256, 0, stream>>>(dstArr, Ee, nE, ebuf, denom, heads);

        aggregate4<<<cdiv(((long)nE * dout) >> 2, 256), 256, 0, stream>>>(
            srcArr, dstArr, Ee, nE, Hbuf, ebuf, agg, doShiftA[l], chShiftA[l], heads);

        if (l < 3)
            bias_elu_cast4<<<cdiv(((long)Nn * dout) >> 2, 256), 256, 0, stream>>>(
                agg, bl[l], xbf, ((long)Nn * dout) >> 2, dout);
        else
            bias_out4<<<cdiv(((long)Nn * dout) >> 2, 256), 256, 0, stream>>>(
                agg, bl[l], (float*)d_out, ((long)Nn * dout) >> 2, dout);
    }
}
